// HierarchicalMILGNN_51049981280322
// MI455X (gfx1250) — compile-verified, matrix operand order checked by reading, not run on hardware
//
#include <hip/hip_runtime.h>
#include <hip/hip_bf16.h>

// ---------------------------------------------------------------------------
// HierarchicalMILGNN for MI455X (gfx1250, wave32, WMMA + async LDS DMA).
//
//   K0: convert+transpose W1, W2 fp32 -> bf16 N-major [N][K] in workspace.
//   K1: per-tile fused GCN layer 1: degrees -> bf16 WMMA GEMM (64x1024x256)
//       with GLOBAL_LOAD_ASYNC_TO_LDS_B128 B-staging -> intra-tile scatter
//       -> relu -> H1 (bf16, ws).
//   K2: per-tile fused GCN layer 2 (64x256x256), A and B staged via async
//       LDS DMA, + scatter + relu + mean pool -> pooled (fp32, ws).
//   K3: 32 blocks of 64 tiles: tile_embs = pooled@Wout+bout (fp32 WMMA),
//       z = relu(emb@Wenc+benc), scores = z@wattn+battn.
//   K4: softmax over 2048 scores, slide = sum(attn*z), logits.
// ---------------------------------------------------------------------------

#define N_TILES 2048
#define N_SUB   64
#define DEG     8
#define NNODES  (N_TILES * N_SUB)          // 131072
#define EPT     (N_SUB * DEG)              // 512 edges per tile
#define F_IN    1024
#define F_HID   256

typedef __attribute__((ext_vector_type(16))) __bf16 v16bf;
typedef __attribute__((ext_vector_type(8)))  __bf16 v8bf;
typedef __attribute__((ext_vector_type(8)))  float  v8f;
typedef __attribute__((ext_vector_type(2)))  float  v2f;

// ---- LDS layout for the two layer kernels (bytes) -------------------------
constexpr int AS_OFF = 0;                        // 64  x 40 bf16 (stride 80B)
constexpr int BT_OFF = AS_OFF + 64 * 40 * 2;     // 5120: 256 x 40 bf16 N-major
constexpr int HS_OFF = BT_OFF + 256 * 40 * 2;    // 25600: 64x256 f32 (g)
constexpr int OS_OFF = HS_OFF + 64 * 256 * 4;    // 91136: 64x256 f32 (scatter)
constexpr int ES_OFF = OS_OFF + 64 * 256 * 4;    // 156672: 512 int src
constexpr int ED_OFF = ES_OFF + EPT * 4;         // 158720: 512 int dst
constexpr int DV_OFF = ED_OFF + EPT * 4;         // 160768: 64 f32 dinv
constexpr int CN_OFF = DV_OFF + 64 * 4;          // 161024: 64 int cnt
constexpr int LDS_LAYER_BYTES = CN_OFF + 64 * 4; // 161280 (< 160KB -> 2 WG/WGP)

// ---- CDNA5 async global -> LDS copy (16B per lane, ASYNCcnt) --------------
__device__ __forceinline__ void async_copy16(void* lds_dst, const void* gsrc) {
  auto ldst = (__attribute__((address_space(3))) char*)lds_dst;
  asm volatile("global_load_async_to_lds_b128 %0, %1, off"
               :: "v"(ldst), "v"(gsrc) : "memory");
}
__device__ __forceinline__ void async_wait0() {
  asm volatile("s_wait_asynccnt 0x0" ::: "memory");
}

// ---- bf16 16x16x32 fragment load from padded-stride-40 LDS ----------------
// ISA 7.12.2: lanes 0-15 -> row=lane, K=0..7 & 16..23; lanes 16-31 ->
// row=lane-16, K=8..15 & 24..31. Same pattern for BsT with N in place of M.
__device__ __forceinline__ v16bf ld_frag40(const __bf16* base, int m0, int lane) {
  const int m  = m0 + (lane & 15);
  const int kb = (lane & 16) ? 8 : 0;
  const __bf16* p = base + m * 40 + kb;
  v8bf lo = *(const v8bf*)(p);
  v8bf hi = *(const v8bf*)(p + 16);
  v16bf r;
#pragma unroll
  for (int j = 0; j < 8; ++j) { r[j] = lo[j]; r[8 + j] = hi[j]; }
  return r;
}

// ---- f32 16x16x4 fragments ------------------------------------------------
__device__ __forceinline__ v2f ld_af32(const float* A, int lda, int m0, int k0, int lane) {
  const int m  = m0 + (lane & 15);
  const int kb = k0 + ((lane & 16) ? 2 : 0);
  v2f a; a[0] = A[m * lda + kb]; a[1] = A[m * lda + kb + 1];
  return a;
}
__device__ __forceinline__ v2f ld_bf32(const float* B, int ldb, int k0, int n0, int lane) {
  const int n  = n0 + (lane & 15);
  const int kb = k0 + ((lane & 16) ? 2 : 0);
  v2f b; b[0] = B[kb * ldb + n]; b[1] = B[(kb + 1) * ldb + n];
  return b;
}

// ---------------------------------------------------------------------------
// Convert + transpose: out[n*K + k] = (bf16) in[k*N + n]   (W is [K][N])
__global__ void cvt_transpose_bf16(const float* __restrict__ in,
                                   __bf16* __restrict__ out, int K, int N) {
  int i = blockIdx.x * blockDim.x + threadIdx.x;
  if (i < K * N) {
    int n = i / K, k = i - n * K;
    out[i] = (__bf16)in[(size_t)k * N + n];
  }
}

// ---------------------------------------------------------------------------
// Fused GCN layer 1: one workgroup per tile, 256 threads = 8 waves.
// Wave w computes rows r0=(w&3)*16, cols c0=(w>>2)*128 (8 col tiles of 16).
__global__ __launch_bounds__(256) void gcn_layer1(
    const float*  __restrict__ x,     // [N, 1024]
    const int*    __restrict__ src, const int* __restrict__ dst,
    const __bf16* __restrict__ W1T,   // [256][1024] bf16, N-major
    const float*  __restrict__ b1,
    __bf16*       __restrict__ H1)    // [N, 256] bf16 out
{
  extern __shared__ char smem[];
  __bf16* As  = (__bf16*)(smem + AS_OFF);
  __bf16* BsT = (__bf16*)(smem + BT_OFF);
  float*  Hs  = (float*)(smem + HS_OFF);
  float*  Os  = (float*)(smem + OS_OFF);
  int*    esrc = (int*)(smem + ES_OFF);
  int*    edst = (int*)(smem + ED_OFF);
  float*  dinv = (float*)(smem + DV_OFF);
  int*    cnt  = (int*)(smem + CN_OFF);

  const int tile = blockIdx.x;
  const int tid  = threadIdx.x;
  const int lane = tid & 31;
  const int wave = tid >> 5;
  const int base = tile * N_SUB;

  // --- degrees (self-loop counts as 1) + stage edge list -------------------
  if (tid < 64) cnt[tid] = 1;
  __syncthreads();
  for (int e = tid; e < EPT; e += 256) {
    int s = src[(size_t)tile * EPT + e] - base;
    int d = dst[(size_t)tile * EPT + e] - base;
    esrc[e] = s; edst[e] = d;
    atomicAdd(&cnt[d], 1);
  }
  __syncthreads();
  if (tid < 64) dinv[tid] = rsqrtf((float)cnt[tid]);

  // --- GEMM: h = x_tile @ W1  (64 x 1024 x 256), bf16 WMMA, f32 acc --------
  const int r0 = (wave & 3) * 16;
  const int c0 = (wave >> 2) * 128;
  v8f acc[8] = {};

  for (int ks = 0; ks < F_IN / 32; ++ks) {
    const int k0 = ks * 32;
    __syncthreads();
    { // B slice via async LDS DMA: BsT[n][0..31] <- W1T[n][k0..k0+31]
      const __bf16* wp = W1T + (size_t)tid * F_IN + k0;
      __bf16* bp = BsT + tid * 40;
#pragma unroll
      for (int q = 0; q < 4; ++q) async_copy16(bp + q * 8, wp + q * 8);
    }
    { // A slice 64x32: fp32 -> bf16 (overlaps with async B in flight)
      const int r = tid >> 2, cc = (tid & 3) * 8;
      const float* xp = x + (size_t)(base + r) * F_IN + k0 + cc;
      __bf16* ap = As + r * 40 + cc;
#pragma unroll
      for (int j = 0; j < 8; ++j) ap[j] = (__bf16)xp[j];
    }
    async_wait0();
    __syncthreads();
    v16bf a = ld_frag40(As, r0, lane);
#pragma unroll
    for (int ct = 0; ct < 8; ++ct) {
      v16bf b = ld_frag40(BsT, c0 + ct * 16, lane);
      acc[ct] = __builtin_amdgcn_wmma_f32_16x16x32_bf16(
          false, a, false, b, (short)0, acc[ct], false, false);
    }
  }

  // --- write g = dinv[row] * h into Hs -------------------------------------
  {
    const int m    = lane & 15;
    const int roff = (lane & 16) ? 8 : 0;
#pragma unroll
    for (int ct = 0; ct < 8; ++ct) {
      const int col = c0 + ct * 16 + m;
#pragma unroll
      for (int j = 0; j < 8; ++j) {
        const int row = r0 + roff + j;
        Hs[row * 256 + col] = acc[ct][j] * dinv[row];
      }
    }
  }
  __syncthreads();

  // --- scatter: Os[d] = g[d] + sum_e g[src_e]; thread owns feature f=tid ---
  for (int i = tid; i < 64 * 256; i += 256) Os[i] = Hs[i];
  __syncthreads();
  {
    const int f = tid;
#pragma unroll 4
    for (int e = 0; e < EPT; ++e) {
      Os[edst[e] * 256 + f] += Hs[esrc[e] * 256 + f];
    }
  }
  __syncthreads();

  // --- epilogue: relu(dinv[d]*Os + b1) -> bf16 -----------------------------
  for (int i = tid; i < 64 * 256; i += 256) {
    const int row = i >> 8, f = i & 255;
    float v = dinv[row] * Os[i] + b1[f];
    H1[(size_t)(base + row) * F_HID + f] = (__bf16)(v > 0.f ? v : 0.f);
  }
}

// ---------------------------------------------------------------------------
// Fused GCN layer 2 + per-tile mean pool. A and B staged via async LDS DMA.
__global__ __launch_bounds__(256) void gcn_layer2(
    const __bf16* __restrict__ H1,    // [N, 256] bf16
    const int*    __restrict__ src, const int* __restrict__ dst,
    const __bf16* __restrict__ W2T,   // [256][256] bf16, N-major
    const float*  __restrict__ b2,
    float*        __restrict__ pooled) // [2048, 256]
{
  extern __shared__ char smem[];
  __bf16* As  = (__bf16*)(smem + AS_OFF);
  __bf16* BsT = (__bf16*)(smem + BT_OFF);
  float*  Hs  = (float*)(smem + HS_OFF);
  float*  Os  = (float*)(smem + OS_OFF);
  int*    esrc = (int*)(smem + ES_OFF);
  int*    edst = (int*)(smem + ED_OFF);
  float*  dinv = (float*)(smem + DV_OFF);
  int*    cnt  = (int*)(smem + CN_OFF);

  const int tile = blockIdx.x;
  const int tid  = threadIdx.x;
  const int lane = tid & 31;
  const int wave = tid >> 5;
  const int base = tile * N_SUB;

  if (tid < 64) cnt[tid] = 1;
  __syncthreads();
  for (int e = tid; e < EPT; e += 256) {
    int s = src[(size_t)tile * EPT + e] - base;
    int d = dst[(size_t)tile * EPT + e] - base;
    esrc[e] = s; edst[e] = d;
    atomicAdd(&cnt[d], 1);
  }
  __syncthreads();
  if (tid < 64) dinv[tid] = rsqrtf((float)cnt[tid]);

  const int r0 = (wave & 3) * 16;
  const int c0 = (wave >> 2) * 128;
  v8f acc[8] = {};

  for (int ks = 0; ks < F_HID / 32; ++ks) {
    const int k0 = ks * 32;
    __syncthreads();
    { // A slice via async DMA (bf16 H1 rows are contiguous)
      const int r = tid >> 2, cc = (tid & 3) * 8;
      async_copy16(As + r * 40 + cc,
                   H1 + (size_t)(base + r) * F_HID + k0 + cc);
    }
    { // B slice via async DMA
      const __bf16* wp = W2T + (size_t)tid * F_HID + k0;
      __bf16* bp = BsT + tid * 40;
#pragma unroll
      for (int q = 0; q < 4; ++q) async_copy16(bp + q * 8, wp + q * 8);
    }
    async_wait0();
    __syncthreads();
    v16bf a = ld_frag40(As, r0, lane);
#pragma unroll
    for (int ct = 0; ct < 8; ++ct) {
      v16bf b = ld_frag40(BsT, c0 + ct * 16, lane);
      acc[ct] = __builtin_amdgcn_wmma_f32_16x16x32_bf16(
          false, a, false, b, (short)0, acc[ct], false, false);
    }
  }

  {
    const int m    = lane & 15;
    const int roff = (lane & 16) ? 8 : 0;
#pragma unroll
    for (int ct = 0; ct < 8; ++ct) {
      const int col = c0 + ct * 16 + m;
#pragma unroll
      for (int j = 0; j < 8; ++j) {
        const int row = r0 + roff + j;
        Hs[row * 256 + col] = acc[ct][j] * dinv[row];
      }
    }
  }
  __syncthreads();

  for (int i = tid; i < 64 * 256; i += 256) Os[i] = Hs[i];
  __syncthreads();
  {
    const int f = tid;
#pragma unroll 4
    for (int e = 0; e < EPT; ++e) {
      Os[edst[e] * 256 + f] += Hs[esrc[e] * 256 + f];
    }
  }
  __syncthreads();

  // relu epilogue in-place, then mean pool over the 64 tile rows
  for (int i = tid; i < 64 * 256; i += 256) {
    const int row = i >> 8, f = i & 255;
    float v = dinv[row] * Os[i] + b2[f];
    Os[i] = v > 0.f ? v : 0.f;
  }
  __syncthreads();
  {
    float s = 0.f;
#pragma unroll 8
    for (int i = 0; i < 64; ++i) s += Os[i * 256 + tid];
    pooled[(size_t)tile * F_HID + tid] = s * (1.0f / 64.0f);
  }
}

// ---------------------------------------------------------------------------
// Head part 1: per 64-tile block. fp32 WMMA (outputs feed d_out directly).
__global__ __launch_bounds__(256) void head_embed(
    const float* __restrict__ pooled,
    const float* __restrict__ Wout, const float* __restrict__ bout,
    const float* __restrict__ Wenc, const float* __restrict__ benc,
    const float* __restrict__ wattn, const float* __restrict__ battn,
    float* __restrict__ emb_out,   // d_out + 2 + 2048
    float* __restrict__ z,         // ws [2048, 256]
    float* __restrict__ scores)    // ws [2048]
{
  extern __shared__ float Ps[];    // 64 x 260 (padded)
  const int blk  = blockIdx.x;
  const int tid  = threadIdx.x;
  const int lane = tid & 31;
  const int wave = tid >> 5;
  const int rb   = blk * 64;

  for (int i = tid; i < 64 * 256; i += 256) {
    const int r = i >> 8, c = i & 255;
    Ps[r * 260 + c] = pooled[(size_t)(rb + r) * 256 + c];
  }
  __syncthreads();

  const int r0 = (wave & 3) * 16;
  const int c0 = (wave >> 2) * 128;

  // emb = pooled @ Wout + bout
  v8f acc[8] = {};
  for (int k0 = 0; k0 < 256; k0 += 4) {
    v2f a = ld_af32(Ps, 260, r0, k0, lane);
#pragma unroll
    for (int ct = 0; ct < 8; ++ct) {
      v2f b = ld_bf32(Wout, 256, k0, c0 + ct * 16, lane);
      acc[ct] = __builtin_amdgcn_wmma_f32_16x16x4_f32(
          false, a, false, b, (short)0, acc[ct], false, false);
    }
  }
  __syncthreads();
  {
    const int m    = lane & 15;
    const int roff = (lane & 16) ? 8 : 0;
#pragma unroll
    for (int ct = 0; ct < 8; ++ct) {
      const int col = c0 + ct * 16 + m;
#pragma unroll
      for (int j = 0; j < 8; ++j) {
        const int row = r0 + roff + j;
        float v = acc[ct][j] + bout[col];
        emb_out[(size_t)(rb + row) * 256 + col] = v;
        Ps[row * 260 + col] = v;
      }
    }
  }
  __syncthreads();

  // z = relu(emb @ Wenc + benc)
  v8f acc2[8] = {};
  for (int k0 = 0; k0 < 256; k0 += 4) {
    v2f a = ld_af32(Ps, 260, r0, k0, lane);
#pragma unroll
    for (int ct = 0; ct < 8; ++ct) {
      v2f b = ld_bf32(Wenc, 256, k0, c0 + ct * 16, lane);
      acc2[ct] = __builtin_amdgcn_wmma_f32_16x16x4_f32(
          false, a, false, b, (short)0, acc2[ct], false, false);
    }
  }
  __syncthreads();
  {
    const int m    = lane & 15;
    const int roff = (lane & 16) ? 8 : 0;
#pragma unroll
    for (int ct = 0; ct < 8; ++ct) {
      const int col = c0 + ct * 16 + m;
#pragma unroll
      for (int j = 0; j < 8; ++j) {
        const int row = r0 + roff + j;
        float v = acc2[ct][j] + benc[col];
        v = v > 0.f ? v : 0.f;
        z[(size_t)(rb + row) * 256 + col] = v;
        Ps[row * 260 + col] = v;
      }
    }
  }
  __syncthreads();

  // scores = z @ wattn + battn (one row per thread 0..63)
  if (tid < 64) {
    float s = battn[0];
#pragma unroll 8
    for (int f = 0; f < 256; ++f) s += Ps[tid * 260 + f] * wattn[f];
    scores[rb + tid] = s;
  }
}

// ---------------------------------------------------------------------------
// Head part 2: softmax over 2048 scores, attn out, slide, logits.
__global__ __launch_bounds__(256) void head_final(
    const float* __restrict__ scores, const float* __restrict__ z,
    const float* __restrict__ Wcls, const float* __restrict__ bcls,
    float* __restrict__ out)   // [0..1] logits, [2..2049] attn
{
  __shared__ float red[256];
  __shared__ float slide[256];
  const int tid = threadIdx.x;

  float m = -3.402823466e+38f;
  for (int t = tid; t < N_TILES; t += 256) m = fmaxf(m, scores[t]);
  red[tid] = m; __syncthreads();
  for (int s = 128; s > 0; s >>= 1) {
    if (tid < s) red[tid] = fmaxf(red[tid], red[tid + s]);
    __syncthreads();
  }
  const float mx = red[0];
  __syncthreads();

  float sum = 0.f;
  for (int t = tid; t < N_TILES; t += 256) sum += __expf(scores[t] - mx);
  red[tid] = sum; __syncthreads();
  for (int s = 128; s > 0; s >>= 1) {
    if (tid < s) red[tid] += red[tid + s];
    __syncthreads();
  }
  const float inv = 1.0f / red[0];

  for (int t = tid; t < N_TILES; t += 256)
    out[2 + t] = __expf(scores[t] - mx) * inv;

  // slide[f] = sum_t attn[t] * z[t][f]  (thread owns feature f = tid)
  float sl = 0.f;
  for (int t = 0; t < N_TILES; ++t) {
    const float a = __expf(scores[t] - mx) * inv;
    sl += a * z[(size_t)t * 256 + tid];
  }
  slide[tid] = sl;
  __syncthreads();

  if (tid < 2) {
    float l = bcls[tid];
#pragma unroll 8
    for (int f = 0; f < 256; ++f) l += slide[f] * Wcls[f * 2 + tid];
    out[tid] = l;
  }
}

// ---------------------------------------------------------------------------
extern "C" void kernel_launch(void* const* d_in, const int* in_sizes, int n_in,
                              void* d_out, int out_size, void* d_ws, size_t ws_size,
                              hipStream_t stream) {
  (void)in_sizes; (void)n_in; (void)out_size; (void)ws_size;

  const float* x     = (const float*)d_in[0];
  const int*   src   = (const int*)d_in[1];
  const int*   dst   = (const int*)d_in[2];
  // d_in[3] node_tile, d_in[4] n_tiles: structure is known statically
  const float* W1    = (const float*)d_in[5];
  const float* b1    = (const float*)d_in[6];
  const float* W2    = (const float*)d_in[7];
  const float* b2    = (const float*)d_in[8];
  const float* Wout  = (const float*)d_in[9];
  const float* bout  = (const float*)d_in[10];
  const float* Wenc  = (const float*)d_in[11];
  const float* benc  = (const float*)d_in[12];
  const float* wattn = (const float*)d_in[13];
  const float* battn = (const float*)d_in[14];
  const float* Wcls  = (const float*)d_in[15];
  const float* bcls  = (const float*)d_in[16];

  // workspace layout
  char* ws = (char*)d_ws;
  __bf16* H1     = (__bf16*)ws;                                  // 64 MB
  size_t  off    = (size_t)NNODES * F_HID * sizeof(__bf16);
  float*  pooled = (float*)(ws + off);  off += (size_t)N_TILES * F_HID * 4;
  float*  z      = (float*)(ws + off);  off += (size_t)N_TILES * F_HID * 4;
  float*  scores = (float*)(ws + off);  off += (size_t)N_TILES * 4;
  __bf16* W1T    = (__bf16*)(ws + off); off += (size_t)F_IN * F_HID * 2;
  __bf16* W2T    = (__bf16*)(ws + off);

  float* out     = (float*)d_out;
  float* emb_out = out + 2 + N_TILES;

  // allow >64KB dynamic LDS (idempotent; not a stream op)
  static bool attr_set = false;
  if (!attr_set) {
    hipFuncSetAttribute((const void*)gcn_layer1,
        hipFuncAttributeMaxDynamicSharedMemorySize, LDS_LAYER_BYTES);
    hipFuncSetAttribute((const void*)gcn_layer2,
        hipFuncAttributeMaxDynamicSharedMemorySize, LDS_LAYER_BYTES);
    hipFuncSetAttribute((const void*)head_embed,
        hipFuncAttributeMaxDynamicSharedMemorySize, 64 * 260 * 4);
    attr_set = true;
  }

  cvt_transpose_bf16<<<(F_IN * F_HID + 255) / 256, 256, 0, stream>>>(
      W1, W1T, F_IN, F_HID);
  cvt_transpose_bf16<<<(F_HID * F_HID + 255) / 256, 256, 0, stream>>>(
      W2, W2T, F_HID, F_HID);

  gcn_layer1<<<N_TILES, 256, LDS_LAYER_BYTES, stream>>>(x, src, dst, W1T, b1, H1);
  gcn_layer2<<<N_TILES, 256, LDS_LAYER_BYTES, stream>>>(H1, src, dst, W2T, b2, pooled);
  head_embed<<<N_TILES / 64, 256, 64 * 260 * 4, stream>>>(
      pooled, Wout, bout, Wenc, benc, wattn, battn, emb_out, z, scores);
  head_final<<<1, 256, 0, stream>>>(scores, z, Wcls, bcls, out);
}
